// GrassmannianLanguageModel_3435973837061
// MI455X (gfx1250) — matrix operand order repaired and checked by reading, not compiled, not run on hardware
//
#include <hip/hip_runtime.h>
#include <hip/hip_bf16.h>

// ---------------- constants (from reference) ----------------
#define BB 8
#define LL 512
#define VV 50257
#define DD 1024
#define NN 1024
#define KK 256
#define TT (BB * LL)        // 4096 rows of X / gi
#define G3 (3 * DD)         // 3072

// ---------------- vector typedefs ----------------
typedef __attribute__((ext_vector_type(16))) __bf16          v16bf;
typedef __attribute__((ext_vector_type(16))) unsigned short  v16u;
typedef __attribute__((ext_vector_type(8)))  unsigned short  v8u;
typedef __attribute__((ext_vector_type(4)))  unsigned short  v4u;
typedef __attribute__((ext_vector_type(8)))  float           v8f;
typedef __attribute__((ext_vector_type(4)))  float           v4f;

// fp32 -> bf16, round-to-nearest-even
static __device__ inline unsigned short f2bf(float f) {
    union { float f; unsigned u; } v; v.f = f;
    unsigned r = v.u + 0x7fffu + ((v.u >> 16) & 1u);
    return (unsigned short)(r >> 16);
}

static __device__ inline v8f wmma_bf16(v16u a, v16u b, v8f c) {
    return __builtin_amdgcn_wmma_f32_16x16x32_bf16(
        false, __builtin_bit_cast(v16bf, a),
        false, __builtin_bit_cast(v16bf, b),
        (short)0, c, false, false);
}

// A-fragment from LDS tile row `q` at k-base `kb` (16x32 bf16 layout, kg = lane>>4)
static __device__ inline v16u a_frag_lds(const unsigned short As[16][1032],
                                         int q, int kb, int kg) {
    v8u lo = *(const v8u*)&As[q][kb + kg * 16];
    v8u hi = *(const v8u*)&As[q][kb + kg * 16 + 8];
    return __builtin_shufflevector(lo, hi, 0,1,2,3,4,5,6,7,8,9,10,11,12,13,14,15);
}

// ---------------- kernel 1: gather token embeddings -> bf16 X ----------------
__global__ void gather_x_kernel(const int* __restrict__ ids,
                                const float* __restrict__ emb_inp,
                                unsigned short* __restrict__ Xb) {
    long i = (long)blockIdx.x * blockDim.x + threadIdx.x;   // over TT*DD
    int row = (int)(i >> 10);
    int col = (int)(i & 1023);
    int tok = ids[row];
    Xb[i] = f2bf(emb_inp[(long)tok * DD + col]);
}

// ---------------- kernel 2: generic fp32 -> bf16 convert ----------------
__global__ void convert_bf16_kernel(const float* __restrict__ src,
                                    unsigned short* __restrict__ dst, long n) {
    long i = (long)blockIdx.x * blockDim.x + threadIdx.x;
    if (i < n) dst[i] = f2bf(src[i]);
}

// ---------------- kernel 3: gi = X (TTxDD) * W_ih^T (G3xDD) + b_ih ----------------
// One WG (4 waves) per 16-row A tile; 192 col-tiles -> 12 groups of 4 per wave.
// k-step is the OUTER loop so the single A-fragment is loaded once and reused
// by 4 WMMAs; only 4 accumulators live -> no spills.
__global__ void gemm_gi_kernel(const unsigned short* __restrict__ Xb,
                               const unsigned short* __restrict__ Wib,
                               const float* __restrict__ b_ih,
                               float* __restrict__ gi) {
    __shared__ unsigned short As[16][1032];   // +8 pad halfs (16B-aligned rows)
    const int tid = threadIdx.x;
    const int t0 = blockIdx.x * 16;

    for (int i = tid; i < 2048; i += 128) {
        int r = i >> 7, c = (i & 127) * 8;
        *(v8u*)&As[r][c] = *(const v8u*)&Xb[(long)(t0 + r) * DD + c];
    }
    __syncthreads();

    const int wave = tid >> 5, lane = tid & 31;
    const int q  = lane & 15;      // A row / B col within tile
    const int kg = lane >> 4;      // K group select

    for (int ctg = 0; ctg < 12; ++ctg) {
        const int g0 = (wave * 48 + ctg * 4) * 16;   // first of 4 col-tiles
        v8f acc0 = {0.f,0.f,0.f,0.f,0.f,0.f,0.f,0.f};
        v8f acc1 = acc0, acc2 = acc0, acc3 = acc0;
        const unsigned short* Wb = &Wib[(long)(g0 + q) * DD + kg * 16];
        #pragma unroll 2
        for (int ks = 0; ks < 32; ++ks) {
            const int kb = ks * 32;
            v16u av = a_frag_lds(As, q, kb, kg);
            acc0 = wmma_bf16(av, *(const v16u*)&Wb[kb],                 acc0);
            acc1 = wmma_bf16(av, *(const v16u*)&Wb[kb + 16 * DD],      acc1);
            acc2 = wmma_bf16(av, *(const v16u*)&Wb[kb + 32 * DD],      acc2);
            acc3 = wmma_bf16(av, *(const v16u*)&Wb[kb + 48 * DD],      acc3);
        }
        v8f* accs[4] = {&acc0, &acc1, &acc2, &acc3};
        for (int cti = 0; cti < 4; ++cti) {
            const int g = g0 + cti * 16;
            const float bias = b_ih[g + q];
            v8f c = *accs[cti];
            for (int r = 0; r < 8; ++r) {
                int m = r + 8 * kg;                  // C layout: M = r + 8*(lane>=16)
                gi[(long)(t0 + m) * G3 + g + q] = c[r] + bias;
            }
        }
    }
}

// ---------------- kernel 4: one GRU step (fp32) ----------------
// thread -> (b = tid&7, d = tid>>3); 8 lanes share each W_hh row (broadcast).
__global__ void gru_step_kernel(const float* __restrict__ gi,
                                const float* __restrict__ h_in,
                                float* __restrict__ h_out,
                                const float* __restrict__ W_hh,
                                const float* __restrict__ b_hh,
                                int t) {
    int tid = blockIdx.x * blockDim.x + threadIdx.x;   // 8192 threads
    int b = tid & 7, d = tid >> 3;

    const v4f* h4  = (const v4f*)(h_in + (long)b * DD);
    const v4f* wr4 = (const v4f*)(W_hh + (long)d * DD);
    const v4f* wz4 = (const v4f*)(W_hh + (long)(d + DD) * DD);
    const v4f* wn4 = (const v4f*)(W_hh + (long)(d + 2 * DD) * DD);

    float hr = 0.f, hz = 0.f, hn = 0.f;
    for (int k = 0; k < DD / 4; ++k) {
        v4f h = h4[k], a = wr4[k], z = wz4[k], n = wn4[k];
        hr += a.x * h.x + a.y * h.y + a.z * h.z + a.w * h.w;
        hz += z.x * h.x + z.y * h.y + z.z * h.z + z.w * h.w;
        hn += n.x * h.x + n.y * h.y + n.z * h.z + n.w * h.w;
    }
    hr += b_hh[d]; hz += b_hh[DD + d]; hn += b_hh[2 * DD + d];

    const float* gir = gi + (long)(b * LL + t) * G3;
    float r = 1.f / (1.f + __expf(-(gir[d] + hr)));
    float z = 1.f / (1.f + __expf(-(gir[DD + d] + hz)));
    float n = tanhf(gir[2 * DD + d] + r * hn);
    float hprev = h_in[(long)b * DD + d];
    h_out[(long)b * DD + d] = (1.f - z) * n + z * hprev;
}

// ---------------- kernel 5: Ut[b][k][n] = bf16( dot(ctx[b], W_A[n*K+k]) + b_A ) ----------------
// One wave per W_A row j (262144 rows); coalesced float4 streaming of the 1 GB matrix.
__global__ void ctx_to_ut_kernel(const float* __restrict__ ctx,
                                 const float* __restrict__ W_A,
                                 const float* __restrict__ b_A,
                                 unsigned short* __restrict__ Ut) {
    int gid = blockIdx.x * blockDim.x + threadIdx.x;
    int j = gid >> 5;                 // wave id == row
    int lane = gid & 31;
    const v4f* wrow = (const v4f*)(W_A + (long)j * DD);
    const v4f* c4   = (const v4f*)ctx;

    float acc[BB];
    for (int b = 0; b < BB; ++b) acc[b] = 0.f;

    for (int it = 0; it < 8; ++it) {
        int k4 = lane + it * 32;
        v4f w = wrow[k4];
        for (int b = 0; b < BB; ++b) {
            v4f h = c4[b * (DD / 4) + k4];
            acc[b] += w.x * h.x + w.y * h.y + w.z * h.z + w.w * h.w;
        }
    }
    for (int off = 16; off > 0; off >>= 1)
        for (int b = 0; b < BB; ++b) acc[b] += __shfl_down(acc[b], off);

    if (lane == 0) {
        float bias = b_A[j];
        int n = j >> 8;               // j = n*K + k
        int k = j & (KK - 1);
        for (int b = 0; b < BB; ++b)
            Ut[(long)b * NN * KK + (long)k * NN + n] = f2bf(acc[b] + bias);
    }
}

// ---------------- kernel 6: logits[b][v] = sum_k ( sum_n emb_geom[v,n]*U[b,n,k] )^2 ----------
// WG (4 waves) per 16-row vocab tile; emb_geom tile converted into LDS once,
// reused across 8 batches. Per batch: k-step outer, 4 col-tiles inner.
__global__ void logits_kernel(const float* __restrict__ emb_geom,
                              const unsigned short* __restrict__ Ut,
                              float* __restrict__ out) {
    __shared__ unsigned short As[16][1032];
    __shared__ float red[4][16];
    const int tid = threadIdx.x;
    const int m0 = blockIdx.x * 16;

    // fill LDS tile (clamp edge rows past V; results masked at store)
    for (int i = tid; i < 4096; i += 128) {
        int r = i >> 8, c = (i & 255) * 4;
        int row = m0 + r; if (row >= VV) row = VV - 1;
        v4f v = *(const v4f*)&emb_geom[(long)row * NN + c];
        v4u u; u.x = f2bf(v.x); u.y = f2bf(v.y); u.z = f2bf(v.z); u.w = f2bf(v.w);
        *(v4u*)&As[r][c] = u;
    }
    __syncthreads();

    const int wave = tid >> 5, lane = tid & 31;
    const int q  = lane & 15;
    const int kg = lane >> 4;
    const int n0 = wave * 64;                 // this wave's 4 col-tiles: n0 .. n0+63

    for (int b = 0; b < BB; ++b) {
        const unsigned short* Ub = &Ut[(long)b * NN * KK + (long)(n0 + q) * NN + kg * 16];
        v8f acc0 = {0.f,0.f,0.f,0.f,0.f,0.f,0.f,0.f};
        v8f acc1 = acc0, acc2 = acc0, acc3 = acc0;
        #pragma unroll 2
        for (int ks = 0; ks < 32; ++ks) {
            const int kb = ks * 32;
            v16u av = a_frag_lds(As, q, kb, kg);
            acc0 = wmma_bf16(av, *(const v16u*)&Ub[kb],               acc0);
            acc1 = wmma_bf16(av, *(const v16u*)&Ub[kb + 16 * NN],    acc1);
            acc2 = wmma_bf16(av, *(const v16u*)&Ub[kb + 32 * NN],    acc2);
            acc3 = wmma_bf16(av, *(const v16u*)&Ub[kb + 48 * NN],    acc3);
        }
        float acc[8];
        for (int r = 0; r < 8; ++r)
            acc[r] = acc0[r]*acc0[r] + acc1[r]*acc1[r] + acc2[r]*acc2[r] + acc3[r]*acc3[r];

        // reduce over N (the 16 lanes of each half-wave)
        for (int m = 1; m <= 8; m <<= 1)
            for (int r = 0; r < 8; ++r) acc[r] += __shfl_xor(acc[r], m);
        if (q == 0)                                    // lanes 0 and 16
            for (int r = 0; r < 8; ++r) red[wave][kg * 8 + r] = acc[r];
        __syncthreads();
        if (tid < 16) {
            float s = red[0][tid] + red[1][tid] + red[2][tid] + red[3][tid];
            int row = m0 + tid;
            if (row < VV) out[(long)b * VV + row] = s;
        }
        __syncthreads();
    }
}

// ---------------- launch ----------------
extern "C" void kernel_launch(void* const* d_in, const int* in_sizes, int n_in,
                              void* d_out, int out_size, void* d_ws, size_t ws_size,
                              hipStream_t stream) {
    (void)in_sizes; (void)n_in; (void)out_size; (void)ws_size;
    const int*   ids      = (const int*)  d_in[0];
    const float* emb_inp  = (const float*)d_in[1];
    const float* W_ih     = (const float*)d_in[2];
    const float* W_hh     = (const float*)d_in[3];
    const float* b_ih     = (const float*)d_in[4];
    const float* b_hh     = (const float*)d_in[5];
    const float* W_A      = (const float*)d_in[6];
    const float* b_A      = (const float*)d_in[7];
    const float* emb_geom = (const float*)d_in[8];
    float* out = (float*)d_out;

    char* ws = (char*)d_ws;
    unsigned short* Xb  = (unsigned short*)(ws);                    //  8 MiB
    unsigned short* Wib = (unsigned short*)(ws + 8388608);          //  6 MiB
    float*          gi  = (float*)(ws + 14680064);                  // 48 MiB
    float*          h0  = (float*)(ws + 65011712);                  // 32 KiB
    float*          h1  = (float*)(ws + 65044480);                  // 32 KiB
    unsigned short* Ut  = (unsigned short*)(ws + 65077248);         //  4 MiB

    gather_x_kernel<<<(TT * DD) / 256, 256, 0, stream>>>(ids, emb_inp, Xb);
    convert_bf16_kernel<<<(G3 * DD) / 256, 256, 0, stream>>>(W_ih, Wib, (long)G3 * DD);
    gemm_gi_kernel<<<TT / 16, 128, 0, stream>>>(Xb, Wib, b_ih, gi);

    hipMemsetAsync(h0, 0, BB * DD * sizeof(float), stream);
    for (int t = 0; t < LL; ++t) {
        float* hin  = (t & 1) ? h1 : h0;
        float* hout = (t & 1) ? h0 : h1;
        gru_step_kernel<<<32, 256, 0, stream>>>(gi, hin, hout, W_hh, b_hh, t);
    }
    // LL even -> final hidden state lands in h0
    ctx_to_ut_kernel<<<(NN * KK) / 4, 128, 0, stream>>>(h0, W_A, b_A, Ut);
    logits_kernel<<<(VV + 15) / 16, 128, 0, stream>>>(emb_geom, Ut, out);
}
